// GraphConvLayer_7894149890339
// MI455X (gfx1250) — matrix operand order repaired and checked by reading, not compiled, not run on hardware
//
#include <hip/hip_runtime.h>
#include <stdint.h>

#define O_N   50000
#define T_N   150000
#define DIN   128
#define HDIM  512
#define DOUT  128
#define K1DIM (3 * DIN)            // 384
#define C2DIM (2 * HDIM + DOUT)    // 1152

typedef __attribute__((ext_vector_type(16))) __bf16 v16bf;
typedef __attribute__((ext_vector_type(8)))  float  v8f;

union ABPack { uint4 u[2]; v16bf v; };

static __device__ __forceinline__ unsigned short f2bf(float f) {
  unsigned int u = __float_as_uint(f);
  u += 0x7FFFu + ((u >> 16) & 1u);   // round-to-nearest-even
  return (unsigned short)(u >> 16);
}

static __device__ __forceinline__ v8f vzero8() {
  v8f z = {0.f, 0.f, 0.f, 0.f, 0.f, 0.f, 0.f, 0.f};
  return z;
}

// D = A(16x32 bf16) x B(32x16 bf16) + C, f32 accum -> v_wmma_f32_16x16x32_bf16
static __device__ __forceinline__ v8f wmma_bf16(uint4 a0, uint4 a1, uint4 b0, uint4 b1, v8f c) {
  ABPack A, B;
  A.u[0] = a0; A.u[1] = a1;
  B.u[0] = b0; B.u[1] = b1;
  return __builtin_amdgcn_wmma_f32_16x16x32_bf16(false, A.v, false, B.v, (short)0, c, false, false);
}

// ---------------------------------------------------------------------------
// K0: zero pool/counts; build transposed bf16 weights Wt[n][k]
// ---------------------------------------------------------------------------
__global__ void prep_kernel(const float* __restrict__ W1, const float* __restrict__ W2,
                            const float* __restrict__ W3, const float* __restrict__ W4,
                            unsigned short* __restrict__ W1t, unsigned short* __restrict__ W2t,
                            unsigned short* __restrict__ W3t, unsigned short* __restrict__ W4t,
                            float* __restrict__ pool, float* __restrict__ counts) {
  const int stride = gridDim.x * blockDim.x;
  const int tid = blockIdx.x * blockDim.x + threadIdx.x;
  for (int i = tid; i < O_N * HDIM; i += stride) pool[i] = 0.f;
  for (int i = tid; i < O_N; i += stride) counts[i] = 0.f;
  for (int i = tid; i < HDIM * K1DIM; i += stride) {     // W1: (384,512) -> W1t[n*384+k]
    int n = i / K1DIM, k = i % K1DIM;
    W1t[i] = f2bf(W1[k * HDIM + n]);
  }
  for (int i = tid; i < C2DIM * HDIM; i += stride) {     // W2: (512,1152) -> W2t[n*512+k]
    int n = i / HDIM, k = i % HDIM;
    W2t[i] = f2bf(W2[k * C2DIM + n]);
  }
  for (int i = tid; i < HDIM * HDIM; i += stride) {      // W3: (512,512)
    int n = i / HDIM, k = i % HDIM;
    W3t[i] = f2bf(W3[k * HDIM + n]);
  }
  for (int i = tid; i < DOUT * HDIM; i += stride) {      // W4: (512,128)
    int n = i / HDIM, k = i % HDIM;
    W4t[i] = f2bf(W4[k * HDIM + n]);
  }
}

// ---------------------------------------------------------------------------
// K1: per-edge degree counts
// ---------------------------------------------------------------------------
__global__ void count_kernel(const int* __restrict__ edges, float* __restrict__ counts) {
  int t = blockIdx.x * blockDim.x + threadIdx.x;
  if (t < T_N) {
    atomicAdd(&counts[edges[2 * t + 0]], 1.f);
    atomicAdd(&counts[edges[2 * t + 1]], 1.f);
  }
}

// ---------------------------------------------------------------------------
// K2: per-32-triple tile: GEMM1 (384->512) + GEMM2 (512->1152) fused.
//     Block = 256 threads = 8 waves. Two 16-row subtiles share every B
//     fragment (2 wmma per 32B of weight traffic from L2).
// ---------------------------------------------------------------------------
__global__ __launch_bounds__(256) void triple_kernel(
    const float* __restrict__ things, const float* __restrict__ preds,
    const int* __restrict__ edges,
    const unsigned short* __restrict__ W1t, const float* __restrict__ b1,
    const unsigned short* __restrict__ W2t, const float* __restrict__ b2,
    float* __restrict__ pool, float* __restrict__ predout)
{
  __shared__ __align__(16) unsigned short sBuf[32 * HDIM];  // 32 KB, reused A->H
  __shared__ int sS[32], sO[32];

  const int tid = threadIdx.x;
  const int t0 = blockIdx.x * 32;

  if (tid < 32) {
    int t = t0 + tid;
    if (t >= T_N) t = T_N - 1;                  // clamp tail: loads stay in-bounds
    sS[tid] = edges[2 * t + 0];
    sO[tid] = edges[2 * t + 1];
  }
  __syncthreads();

  // Phase A: build bf16 triple tile [things[s] | predicates | things[o]] (32x384)
  for (int i = tid; i < 32 * K1DIM; i += 256) {
    int m = i / K1DIM, c = i % K1DIM;
    int t = t0 + m; if (t >= T_N) t = T_N - 1;
    float v;
    if (c < DIN)            v = things[(size_t)sS[m] * DIN + c];
    else if (c < 2 * DIN)   v = preds[(size_t)t * DIN + (c - DIN)];
    else                    v = things[(size_t)sO[m] * DIN + (c - 2 * DIN)];
    sBuf[i] = f2bf(v);
  }
  __syncthreads();

  const int wave = tid >> 5;
  const int lane = tid & 31;
  const int half = lane >> 4;     // K-half selector for A/B fragments
  const int nn   = lane & 15;     // row (A) / column (B,D) within tile

  // ---- GEMM1: h = relu(A @ W1 + b1); wave covers 64 cols x 32 rows ----
  v8f acc1[2][4];
  #pragma unroll
  for (int s = 0; s < 2; ++s)
    #pragma unroll
    for (int j = 0; j < 4; ++j) acc1[s][j] = vzero8();

  #pragma unroll
  for (int k0 = 0; k0 < K1DIM; k0 += 32) {
    uint4 a00 = *(const uint4*)&sBuf[(nn)      * K1DIM + k0 + 8 * half];
    uint4 a01 = *(const uint4*)&sBuf[(nn)      * K1DIM + k0 + 16 + 8 * half];
    uint4 a10 = *(const uint4*)&sBuf[(16 + nn) * K1DIM + k0 + 8 * half];
    uint4 a11 = *(const uint4*)&sBuf[(16 + nn) * K1DIM + k0 + 16 + 8 * half];
    #pragma unroll
    for (int j = 0; j < 4; ++j) {
      int ncol = wave * 64 + j * 16 + nn;
      const unsigned short* bp = &W1t[(size_t)ncol * K1DIM + k0 + 8 * half];
      uint4 b0 = *(const uint4*)(bp);
      uint4 bx = *(const uint4*)(bp + 16);
      acc1[0][j] = wmma_bf16(a00, a01, b0, bx, acc1[0][j]);
      acc1[1][j] = wmma_bf16(a10, a11, b0, bx, acc1[1][j]);
    }
  }
  __syncthreads();   // all waves done reading A region before overwriting with h

  #pragma unroll
  for (int s = 0; s < 2; ++s)
    #pragma unroll
    for (int j = 0; j < 4; ++j) {
      int ncol = wave * 64 + j * 16 + nn;
      float bias = b1[ncol];
      #pragma unroll
      for (int r = 0; r < 8; ++r) {
        float v = acc1[s][j][r] + bias;
        v = v > 0.f ? v : 0.f;
        sBuf[(16 * s + r + 8 * half) * HDIM + ncol] = f2bf(v);  // m = r + 8*(lane/16)
      }
    }
  __syncthreads();

  // ---- GEMM2: c2 = relu(h @ W2 + b2); 1152 cols = 8 waves x 9 tiles ----
  v8f acc2[2][9];
  #pragma unroll
  for (int s = 0; s < 2; ++s)
    #pragma unroll
    for (int j = 0; j < 9; ++j) acc2[s][j] = vzero8();

  for (int k0 = 0; k0 < HDIM; k0 += 32) {
    uint4 a00 = *(const uint4*)&sBuf[(nn)      * HDIM + k0 + 8 * half];
    uint4 a01 = *(const uint4*)&sBuf[(nn)      * HDIM + k0 + 16 + 8 * half];
    uint4 a10 = *(const uint4*)&sBuf[(16 + nn) * HDIM + k0 + 8 * half];
    uint4 a11 = *(const uint4*)&sBuf[(16 + nn) * HDIM + k0 + 16 + 8 * half];
    #pragma unroll
    for (int j = 0; j < 9; ++j) {
      int ncol = (wave * 9 + j) * 16 + nn;
      const unsigned short* bp = &W2t[(size_t)ncol * HDIM + k0 + 8 * half];
      uint4 b0 = *(const uint4*)(bp);
      uint4 bx = *(const uint4*)(bp + 16);
      acc2[0][j] = wmma_bf16(a00, a01, b0, bx, acc2[0][j]);
      acc2[1][j] = wmma_bf16(a10, a11, b0, bx, acc2[1][j]);
    }
  }

  // Epilogue: subjects -> pool[s], predicates -> d_out, objects -> pool[o]
  #pragma unroll
  for (int s = 0; s < 2; ++s)
    #pragma unroll
    for (int j = 0; j < 9; ++j) {
      int gtile = wave * 9 + j;
      int gcol = gtile * 16 + nn;
      float bias = b2[gcol];
      #pragma unroll
      for (int r = 0; r < 8; ++r) {
        int m = 16 * s + r + 8 * half;
        if (t0 + m >= T_N) continue;            // tail guard (stores only)
        float v = acc2[s][j][r] + bias;
        v = v > 0.f ? v : 0.f;
        if (gtile < 32) {                       // new_subjects: cols [0,512)
          atomicAdd(&pool[(size_t)sS[m] * HDIM + gcol], v);
        } else if (gtile < 40) {                // new_predicates: cols [512,640)
          predout[(size_t)(t0 + m) * DOUT + (gcol - HDIM)] = v;
        } else {                                // new_objects: cols [640,1152)
          atomicAdd(&pool[(size_t)sO[m] * HDIM + (gcol - (HDIM + DOUT))], v);
        }
      }
    }
}

// ---------------------------------------------------------------------------
// K3: per-32-object tile: normalize pool, GEMM3 (512->512) + GEMM4 (512->128)
// ---------------------------------------------------------------------------
__global__ __launch_bounds__(256) void node_kernel(
    const float* __restrict__ pool, const float* __restrict__ counts,
    const unsigned short* __restrict__ W3t, const float* __restrict__ b3,
    const unsigned short* __restrict__ W4t, const float* __restrict__ b4,
    float* __restrict__ out_things)
{
  __shared__ __align__(16) unsigned short sBuf[32 * HDIM];  // 32 KB, reused A->H
  __shared__ float sInv[32];

  const int tid = threadIdx.x;
  const int t0 = blockIdx.x * 32;

  if (tid < 32) {
    int t = t0 + tid; if (t >= O_N) t = O_N - 1;
    float c = counts[t];
    sInv[tid] = 1.f / (c > 1.f ? c : 1.f);
  }
  __syncthreads();

  for (int i = tid; i < 32 * HDIM; i += 256) {
    int m = i / HDIM;
    int t = t0 + m; if (t >= O_N) t = O_N - 1;
    sBuf[i] = f2bf(pool[(size_t)t * HDIM + (i % HDIM)] * sInv[m]);
  }
  __syncthreads();

  const int wave = tid >> 5;
  const int lane = tid & 31;
  const int half = lane >> 4;
  const int nn   = lane & 15;

  // ---- GEMM3: g = relu(pool @ W3 + b3); 512 cols = 8 waves x 4 tiles ----
  v8f acc3[2][4];
  #pragma unroll
  for (int s = 0; s < 2; ++s)
    #pragma unroll
    for (int j = 0; j < 4; ++j) acc3[s][j] = vzero8();

  for (int k0 = 0; k0 < HDIM; k0 += 32) {
    uint4 a00 = *(const uint4*)&sBuf[(nn)      * HDIM + k0 + 8 * half];
    uint4 a01 = *(const uint4*)&sBuf[(nn)      * HDIM + k0 + 16 + 8 * half];
    uint4 a10 = *(const uint4*)&sBuf[(16 + nn) * HDIM + k0 + 8 * half];
    uint4 a11 = *(const uint4*)&sBuf[(16 + nn) * HDIM + k0 + 16 + 8 * half];
    #pragma unroll
    for (int j = 0; j < 4; ++j) {
      int ncol = wave * 64 + j * 16 + nn;
      const unsigned short* bp = &W3t[(size_t)ncol * HDIM + k0 + 8 * half];
      uint4 b0 = *(const uint4*)(bp);
      uint4 bx = *(const uint4*)(bp + 16);
      acc3[0][j] = wmma_bf16(a00, a01, b0, bx, acc3[0][j]);
      acc3[1][j] = wmma_bf16(a10, a11, b0, bx, acc3[1][j]);
    }
  }
  __syncthreads();   // done reading sBuf as A

  #pragma unroll
  for (int s = 0; s < 2; ++s)
    #pragma unroll
    for (int j = 0; j < 4; ++j) {
      int ncol = wave * 64 + j * 16 + nn;
      float bias = b3[ncol];
      #pragma unroll
      for (int r = 0; r < 8; ++r) {
        float v = acc3[s][j][r] + bias;
        v = v > 0.f ? v : 0.f;
        sBuf[(16 * s + r + 8 * half) * HDIM + ncol] = f2bf(v);
      }
    }
  __syncthreads();

  // ---- GEMM4: out = relu(g @ W4 + b4); 128 cols = 8 waves x 1 tile ----
  v8f acc4[2];
  acc4[0] = vzero8(); acc4[1] = vzero8();
  for (int k0 = 0; k0 < HDIM; k0 += 32) {
    uint4 a00 = *(const uint4*)&sBuf[(nn)      * HDIM + k0 + 8 * half];
    uint4 a01 = *(const uint4*)&sBuf[(nn)      * HDIM + k0 + 16 + 8 * half];
    uint4 a10 = *(const uint4*)&sBuf[(16 + nn) * HDIM + k0 + 8 * half];
    uint4 a11 = *(const uint4*)&sBuf[(16 + nn) * HDIM + k0 + 16 + 8 * half];
    int ncol = wave * 16 + nn;
    const unsigned short* bp = &W4t[(size_t)ncol * HDIM + k0 + 8 * half];
    uint4 b0 = *(const uint4*)(bp);
    uint4 bx = *(const uint4*)(bp + 16);
    acc4[0] = wmma_bf16(a00, a01, b0, bx, acc4[0]);
    acc4[1] = wmma_bf16(a10, a11, b0, bx, acc4[1]);
  }
  {
    int ncol = wave * 16 + nn;
    float bias = b4[ncol];
    #pragma unroll
    for (int s = 0; s < 2; ++s)
      #pragma unroll
      for (int r = 0; r < 8; ++r) {
        int m = 16 * s + r + 8 * half;
        if (t0 + m >= O_N) continue;            // tail guard
        float v = acc4[s][r] + bias;
        v = v > 0.f ? v : 0.f;
        out_things[(size_t)(t0 + m) * DOUT + ncol] = v;
      }
  }
}

// ---------------------------------------------------------------------------
extern "C" void kernel_launch(void* const* d_in, const int* in_sizes, int n_in,
                              void* d_out, int out_size, void* d_ws, size_t ws_size,
                              hipStream_t stream) {
  const float* things = (const float*)d_in[0];
  const float* preds  = (const float*)d_in[1];
  const int*   edges  = (const int*)d_in[2];
  const float* W1 = (const float*)d_in[3];
  const float* b1 = (const float*)d_in[4];
  const float* W2 = (const float*)d_in[5];
  const float* b2 = (const float*)d_in[6];
  const float* W3 = (const float*)d_in[7];
  const float* b3 = (const float*)d_in[8];
  const float* W4 = (const float*)d_in[9];
  const float* b4 = (const float*)d_in[10];

  float* out_things = (float*)d_out;                           // (O, 128)
  float* out_preds  = out_things + (size_t)O_N * DOUT;         // (T, 128)

  char* ws = (char*)d_ws;
  size_t off = 0;
  auto take = [&](size_t bytes) -> char* {
    char* p = ws + off;
    off = (off + bytes + 255) & ~(size_t)255;
    return p;
  };
  float*          pool   = (float*)take((size_t)O_N * HDIM * sizeof(float));   // 102.4 MB
  float*          counts = (float*)take((size_t)O_N * sizeof(float));
  unsigned short* W1t    = (unsigned short*)take((size_t)HDIM * K1DIM * 2);
  unsigned short* W2t    = (unsigned short*)take((size_t)C2DIM * HDIM * 2);
  unsigned short* W3t    = (unsigned short*)take((size_t)HDIM * HDIM * 2);
  unsigned short* W4t    = (unsigned short*)take((size_t)DOUT * HDIM * 2);

  prep_kernel<<<2048, 256, 0, stream>>>(W1, W2, W3, W4, W1t, W2t, W3t, W4t, pool, counts);
  count_kernel<<<(T_N + 255) / 256, 256, 0, stream>>>(edges, counts);
  triple_kernel<<<(T_N + 31) / 32, 256, 0, stream>>>(things, preds, edges,
                                                     W1t, b1, W2t, b2, pool, out_preds);
  node_kernel<<<(O_N + 31) / 32, 256, 0, stream>>>(pool, counts, W3t, b3, W4t, b4, out_things);
}
